// AutoIntPlus_76630806495343
// MI455X (gfx1250) — matrix operand order repaired and checked by reading, not compiled, not run on hardware
//
#include <hip/hip_runtime.h>
#include <math.h>

typedef __attribute__((ext_vector_type(16))) _Float16 v16h;
typedef __attribute__((ext_vector_type(8)))  float    v8f;

#define BATCH 16384
#define NF    39          // fields
#define FP    48          // fields padded to 3x16 WMMA tiles
#define DH    32          // attention dim (2 N-tiles)
#define KATT  64          // att K padded to 2x32
#define LDH   1248        // NF*DH flattened feature dim
#define VOCAB 10000

// gfx1250 async global->LDS copy, guarded so the file still compiles if this
// toolchain doesn't expose the builtins (falls back to plain LDS staging).
#if defined(__has_builtin)
#if __has_builtin(__builtin_amdgcn_global_load_async_to_lds_b128) && \
    __has_builtin(__builtin_amdgcn_s_wait_asynccnt)
#define HAVE_ASYNC_LDS 1
#endif
#endif

#ifdef HAVE_ASYNC_LDS
// builtin signature (from hipcc diagnostic): (v4i AS1*, v4i AS3*, imm int, imm int)
typedef int async_v4i __attribute__((vector_size(4 * sizeof(int))));
typedef __attribute__((address_space(1))) async_v4i* async_gp128;
typedef __attribute__((address_space(3))) async_v4i* async_lp128;
__device__ __forceinline__ void async_cp_b128(const void* g, void* l) {
  // generic->AS casts must go through integers; LDS offset = addr[31:0]
  __builtin_amdgcn_global_load_async_to_lds_b128(
      (async_gp128)(unsigned long long)(uintptr_t)g,
      (async_lp128)(unsigned int)(uintptr_t)l, 0, 0);
}
#endif

// ---------------------------------------------------------------------------
// WMMA helpers (CDNA5 wave32, V_WMMA_F32_16X16X32_F16)
// A layout: lane = row (m = lane&15), elems 0..15 = K[kh..kh+15], kh=(lane>>4)*16
// B layout: lane = col (n = lane&15), elems 0..15 = K[kh..kh+15], kh=(lane>>4)*16
// C layout: vgpr r, lane l -> (m = r + 8*(l>>4), n = l&15)
// ---------------------------------------------------------------------------
__device__ __forceinline__ v16h load_a_tile(const _Float16* base, int row0, int ldk, int k0) {
  const int lane = threadIdx.x & 31;
  const int m    = lane & 15;
  const int kh   = (lane >> 4) << 4;
  const _Float16* p = base + (size_t)(row0 + m) * ldk + k0;
  v16h a;
#pragma unroll
  for (int j = 0; j < 16; ++j) {
    a[j]     = p[kh + j];
  }
  return a;
}

__device__ __forceinline__ v16h load_b_tile(const _Float16* base, int col0, int ldk, int k0) {
  const int lane = threadIdx.x & 31;
  const int n    = lane & 15;
  const int kh   = (lane >> 4) << 4;
  const _Float16* p = base + (size_t)(col0 + n) * ldk + k0 + kh;
  v16h b;
#pragma unroll
  for (int j = 0; j < 16; ++j) b[j] = p[j];
  return b;
}

__device__ __forceinline__ v8f wmma16(v16h a, v16h b, v8f c) {
  return __builtin_amdgcn_wmma_f32_16x16x32_f16(false, a, false, b, (short)0, c, false, false);
}

// ---------------------------------------------------------------------------
// Weight preconversion: dst[n*padK + k] = f16(src[k*N + n]), zero pad k >= K
// ---------------------------------------------------------------------------
__global__ void convw_kernel(const float* __restrict__ src, _Float16* __restrict__ dst,
                             int K, int N, int padK) {
  int i = blockIdx.x * blockDim.x + threadIdx.x;
  if (i >= N * padK) return;
  int n = i / padK, k = i % padK;
  dst[(size_t)n * padK + k] = (k < K) ? (_Float16)src[(size_t)k * N + n] : (_Float16)0.f;
}

// ---------------------------------------------------------------------------
// Fused: embedding gather + 3 attention layers. 2 waves/block, 1 sample/wave,
// everything stays in LDS; writes final e (flattened, f16) to h_out[B][1248].
// ---------------------------------------------------------------------------
__global__ __launch_bounds__(64)
void attn_kernel(const int* __restrict__ x, const float* __restrict__ emb,
                 const _Float16* __restrict__ wproj,   // 12 x [32][32] f16 (n-major)
                 _Float16* __restrict__ h_out) {
  __shared__ _Float16 bufE[2][FP * DH];     // e (current layer input)
  __shared__ _Float16 bufQ[2][FP * DH];     // q, later reused for new e
  __shared__ _Float16 kbuf_s[2][FP * DH];   // k
  __shared__ _Float16 vT_s[2][DH * KATT];   // v transposed: [n][field], padded to 64
  __shared__ float    sc_s[2][FP * FP];     // raw scores f32
  __shared__ _Float16 attA_s[2][FP * KATT]; // softmax(att), row-major, K padded to 64

  const int wid  = threadIdx.x >> 5;
  const int lane = threadIdx.x & 31;
  const int s    = blockIdx.x * 2 + wid;
  const int n    = lane & 15;
  const int mh   = (lane >> 4) << 3;

  _Float16* cur  = bufE[wid];
  _Float16* oth  = bufQ[wid];
  _Float16* kbuf = kbuf_s[wid];
  _Float16* vT   = vT_s[wid];
  float*    sc   = sc_s[wid];
  _Float16* attA = attA_s[wid];

  // ---- embedding gather (layer0 K padded 16->32 with zeros) ----
  for (int i = lane; i < FP * DH; i += 32) cur[i] = (_Float16)0.f;
  __syncthreads();
  for (int i = lane; i < NF * 16; i += 32) {
    int f = i >> 4, d = i & 15;
    int tok = x[(size_t)s * NF + f] + f * VOCAB;
    cur[f * DH + d] = (_Float16)emb[(size_t)tok * 16 + d];
  }
  __syncthreads();

  for (int layer = 0; layer < 3; ++layer) {
    const _Float16* wq = wproj + (size_t)(layer * 4 + 0) * 1024;
    const _Float16* wk = wproj + (size_t)(layer * 4 + 1) * 1024;
    const _Float16* wv = wproj + (size_t)(layer * 4 + 2) * 1024;
    const _Float16* wr = wproj + (size_t)(layer * 4 + 3) * 1024;

    // ---- q = e@Wq -> oth,  k = e@Wk -> kbuf ----
    for (int mt = 0; mt < 3; ++mt) {
      v16h a = load_a_tile(cur, mt * 16, DH, 0);
      for (int nt = 0; nt < 2; ++nt) {
        v8f cq = {}; cq = wmma16(a, load_b_tile(wq, nt * 16, DH, 0), cq);
        v8f ck = {}; ck = wmma16(a, load_b_tile(wk, nt * 16, DH, 0), ck);
#pragma unroll
        for (int r = 0; r < 8; ++r) {
          int row = mt * 16 + mh + r;
          oth [row * DH + nt * 16 + n] = (_Float16)cq[r];
          kbuf[row * DH + nt * 16 + n] = (_Float16)ck[r];
        }
      }
    }
    __syncthreads();

    // ---- scores = q @ k^T  (B-col of k^T == k row => same addressing) ----
    for (int mt = 0; mt < 3; ++mt) {
      v16h a = load_a_tile(oth, mt * 16, DH, 0);
      for (int nt = 0; nt < 3; ++nt) {
        v8f c = {}; c = wmma16(a, load_b_tile(kbuf, nt * 16, DH, 0), c);
#pragma unroll
        for (int r = 0; r < 8; ++r)
          sc[(mt * 16 + mh + r) * FP + nt * 16 + n] = c[r];
      }
    }
    __syncthreads();

    // ---- softmax rows (mask padded cols >= NF, zero padded rows) ----
    for (int row = lane; row < FP; row += 32) {
      if (row < NF) {
        float mx = -1e30f;
        for (int c2 = 0; c2 < NF; ++c2) mx = fmaxf(mx, sc[row * FP + c2]);
        float sum = 0.f;
        for (int c2 = 0; c2 < NF; ++c2) {
          float e = expf(sc[row * FP + c2] - mx);
          sc[row * FP + c2] = e; sum += e;
        }
        float inv = 1.f / sum;
        for (int c2 = 0; c2 < KATT; ++c2)
          attA[row * KATT + c2] = (c2 < NF) ? (_Float16)(sc[row * FP + c2] * inv)
                                            : (_Float16)0.f;
      } else {
        for (int c2 = 0; c2 < KATT; ++c2) attA[row * KATT + c2] = (_Float16)0.f;
      }
    }

    // ---- v = e@Wv, stored transposed into vT (zero-init: padded K & stale LDS) ----
    for (int i = lane; i < DH * KATT; i += 32) vT[i] = (_Float16)0.f;
    __syncthreads();
    for (int mt = 0; mt < 3; ++mt) {
      v16h a = load_a_tile(cur, mt * 16, DH, 0);
      for (int nt = 0; nt < 2; ++nt) {
        v8f c = {}; c = wmma16(a, load_b_tile(wv, nt * 16, DH, 0), c);
#pragma unroll
        for (int r = 0; r < 8; ++r)
          vT[(nt * 16 + n) * KATT + mt * 16 + mh + r] = (_Float16)c[r];
      }
    }
    __syncthreads();

    // ---- out = relu(e@Wr + att@v) -> oth (overwrites q, no longer needed) ----
    for (int mt = 0; mt < 3; ++mt) {
      v16h ae = load_a_tile(cur,  mt * 16, DH,   0);
      v16h a0 = load_a_tile(attA, mt * 16, KATT, 0);
      v16h a1 = load_a_tile(attA, mt * 16, KATT, 32);
      for (int nt = 0; nt < 2; ++nt) {
        v8f c = {};
        c = wmma16(ae, load_b_tile(wr, nt * 16, DH,   0),  c);
        c = wmma16(a0, load_b_tile(vT, nt * 16, KATT, 0),  c);
        c = wmma16(a1, load_b_tile(vT, nt * 16, KATT, 32), c);
#pragma unroll
        for (int r = 0; r < 8; ++r) {
          int row = mt * 16 + mh + r;
          float v = fmaxf(c[r], 0.f);
          oth[row * DH + nt * 16 + n] = (row < NF) ? (_Float16)v : (_Float16)0.f;
        }
      }
    }
    __syncthreads();
    _Float16* t = cur; cur = oth; oth = t;
  }

  // ---- flatten: e[39][32] row-major == reshape(F*DH) ----
  for (int i = lane; i < NF * DH; i += 32)
    h_out[(size_t)s * LDH + i] = cur[i];
}

// ---------------------------------------------------------------------------
// Fused MLP 1248->128->64->32->1 + sigmoid. One wave per 16-sample tile,
// 4 waves per block. wt2/wt3 async-prefetched into LDS (overlaps stage 1),
// shared by all 4 waves. K-outer loops keep 8/4/2 live WMMA accumulators.
// ---------------------------------------------------------------------------
__global__ __launch_bounds__(128)
void mlp_kernel(const _Float16* __restrict__ h,
                const _Float16* __restrict__ wt1, const float* __restrict__ b1,
                const _Float16* __restrict__ wt2, const float* __restrict__ b2,
                const _Float16* __restrict__ wt3, const float* __restrict__ b3,
                const float* __restrict__ w4,  const float* __restrict__ b4,
                float* __restrict__ out) {
  __shared__ _Float16 h1s[4][16 * 128];
  __shared__ _Float16 h2s[4][16 * 64];
  __shared__ _Float16 h3s[4][16 * 32];
  __shared__ _Float16 wt2s[64 * 128];   // 16 KB, block-shared
  __shared__ _Float16 wt3s[32 * 64];    //  4 KB, block-shared

  const int tid  = threadIdx.x;
  const int wid  = tid >> 5;
  const int lane = tid & 31;
  const int n    = lane & 15;
  const int mh   = (lane >> 4) << 3;
  const int row0 = (blockIdx.x * 4 + wid) * 16;

  _Float16* h1 = h1s[wid];
  _Float16* h2 = h2s[wid];
  _Float16* h3 = h3s[wid];

  // ---- kick off async staging of wt2/wt3 into LDS; completes under stage 1 ----
#ifdef HAVE_ASYNC_LDS
  for (int i = tid; i < (64 * 128) / 8; i += 128)     // 1024 x b128
    async_cp_b128(wt2 + (size_t)i * 8, wt2s + (size_t)i * 8);
  for (int i = tid; i < (32 * 64) / 8; i += 128)      // 256 x b128
    async_cp_b128(wt3 + (size_t)i * 8, wt3s + (size_t)i * 8);
#else
  for (int i = tid; i < 64 * 128; i += 128) wt2s[i] = wt2[i];
  for (int i = tid; i < 32 * 64;  i += 128) wt3s[i] = wt3[i];
#endif

  // ---- stage 1: [16,1248] x [1248,128] ----
  {
    v8f acc[8];
#pragma unroll
    for (int nt = 0; nt < 8; ++nt) {
      float bias = b1[nt * 16 + n];
#pragma unroll
      for (int j = 0; j < 8; ++j) acc[nt][j] = bias;
    }
    for (int kt = 0; kt < 39; ++kt) {
      if (kt + 1 < 39)   // gfx1250 global_prefetch_b8 on next A panel
        __builtin_prefetch(h + (size_t)(row0 + (lane & 15)) * LDH + (kt + 1) * 32, 0, 1);
      v16h a = load_a_tile(h, row0, LDH, kt * 32);
#pragma unroll
      for (int nt = 0; nt < 8; ++nt)
        acc[nt] = wmma16(a, load_b_tile(wt1, nt * 16, LDH, kt * 32), acc[nt]);
    }
#pragma unroll
    for (int nt = 0; nt < 8; ++nt)
#pragma unroll
      for (int r = 0; r < 8; ++r)
        h1[(mh + r) * 128 + nt * 16 + n] = (_Float16)fmaxf(acc[nt][r], 0.f);
  }
#ifdef HAVE_ASYNC_LDS
  __builtin_amdgcn_s_wait_asynccnt(0);
#endif
  __syncthreads();

  // ---- stage 2: [16,128] x [128,64], B from LDS ----
  {
    v8f acc[4];
#pragma unroll
    for (int nt = 0; nt < 4; ++nt) {
      float bias = b2[nt * 16 + n];
#pragma unroll
      for (int j = 0; j < 8; ++j) acc[nt][j] = bias;
    }
#pragma unroll
    for (int kt = 0; kt < 4; ++kt) {
      v16h a = load_a_tile(h1, 0, 128, kt * 32);
#pragma unroll
      for (int nt = 0; nt < 4; ++nt)
        acc[nt] = wmma16(a, load_b_tile(wt2s, nt * 16, 128, kt * 32), acc[nt]);
    }
#pragma unroll
    for (int nt = 0; nt < 4; ++nt)
#pragma unroll
      for (int r = 0; r < 8; ++r)
        h2[(mh + r) * 64 + nt * 16 + n] = (_Float16)fmaxf(acc[nt][r], 0.f);
  }
  __syncthreads();

  // ---- stage 3: [16,64] x [64,32], B from LDS ----
  {
    v8f acc[2];
#pragma unroll
    for (int nt = 0; nt < 2; ++nt) {
      float bias = b3[nt * 16 + n];
#pragma unroll
      for (int j = 0; j < 8; ++j) acc[nt][j] = bias;
    }
#pragma unroll
    for (int kt = 0; kt < 2; ++kt) {
      v16h a = load_a_tile(h2, 0, 64, kt * 32);
#pragma unroll
      for (int nt = 0; nt < 2; ++nt)
        acc[nt] = wmma16(a, load_b_tile(wt3s, nt * 16, 64, kt * 32), acc[nt]);
    }
#pragma unroll
    for (int nt = 0; nt < 2; ++nt)
#pragma unroll
      for (int r = 0; r < 8; ++r)
        h3[(mh + r) * 32 + nt * 16 + n] = (_Float16)fmaxf(acc[nt][r], 0.f);
  }
  __syncthreads();

  // ---- stage 4: [16,32] . w4 + b4, sigmoid ----
  if (lane < 16) {
    float s = b4[0];
    for (int k = 0; k < 32; ++k) s += (float)h3[lane * 32 + k] * w4[k];
    out[row0 + lane] = 1.f / (1.f + expf(-s));
  }
}

// ---------------------------------------------------------------------------
extern "C" void kernel_launch(void* const* d_in, const int* in_sizes, int n_in,
                              void* d_out, int out_size, void* d_ws, size_t ws_size,
                              hipStream_t stream) {
  (void)in_sizes; (void)n_in; (void)out_size; (void)ws_size;

  const int*   x   = (const int*)d_in[0];
  const float* emb = (const float*)d_in[1];
  const float* wattn[12];
  for (int i = 0; i < 12; ++i) wattn[i] = (const float*)d_in[2 + i];
  const float* w1 = (const float*)d_in[14]; const float* b1 = (const float*)d_in[15];
  const float* w2 = (const float*)d_in[16]; const float* b2 = (const float*)d_in[17];
  const float* w3 = (const float*)d_in[18]; const float* b3 = (const float*)d_in[19];
  const float* w4 = (const float*)d_in[20]; const float* b4 = (const float*)d_in[21];

  // workspace layout (f16 elements)
  _Float16* ws    = (_Float16*)d_ws;
  _Float16* wproj = ws;                          // 12 * 1024
  _Float16* wt1   = wproj + 12 * 1024;           // 128 * 1248
  _Float16* wt2   = wt1 + (size_t)128 * LDH;     // 64 * 128
  _Float16* wt3   = wt2 + 64 * 128;              // 32 * 64
  _Float16* hbuf  = wt3 + 32 * 64;               // BATCH * 1248

  // weight conversion/transposition to f16 n-major (zero-padded K for layer 0)
  for (int l = 0; l < 3; ++l) {
    int K = (l == 0) ? 16 : 32;
    for (int p = 0; p < 4; ++p)
      convw_kernel<<<(32 * 32 + 255) / 256, 256, 0, stream>>>(
          wattn[l * 4 + p], wproj + (size_t)(l * 4 + p) * 1024, K, 32, 32);
  }
  convw_kernel<<<((128 * LDH) + 255) / 256, 256, 0, stream>>>(w1, wt1, LDH, 128, LDH);
  convw_kernel<<<((64 * 128) + 255) / 256, 256, 0, stream>>>(w2, wt2, 128, 64, 128);
  convw_kernel<<<((32 * 64) + 255) / 256, 256, 0, stream>>>(w3, wt3, 64, 32, 64);

  attn_kernel<<<BATCH / 2, 64, 0, stream>>>(x, emb, wproj, hbuf);
  mlp_kernel<<<BATCH / 64, 128, 0, stream>>>(hbuf, wt1, b1, wt2, b2, wt3, b3,
                                             w4, b4, (float*)d_out);
}
